// CausalSelfAttention_48747878809753
// MI455X (gfx1250) — matrix operand order repaired and checked by reading, not compiled, not run on hardware
//
#include <hip/hip_runtime.h>
#include <hip/hip_bf16.h>

// ---------------------------------------------------------------------------
// Types for gfx1250 WMMA (wave32)
// ---------------------------------------------------------------------------
typedef __bf16 bf16_t;
typedef __attribute__((ext_vector_type(16))) __bf16 v16bf;
typedef __attribute__((ext_vector_type(8)))  __bf16 v8bf;
typedef __attribute__((ext_vector_type(8)))  float  v8f;

static __device__ __forceinline__ v8f vzero8f() {
    v8f z = {0.f, 0.f, 0.f, 0.f, 0.f, 0.f, 0.f, 0.f};
    return z;
}

static __device__ __forceinline__ v16bf combine8(v8bf lo, v8bf hi) {
    return __builtin_shufflevector(lo, hi, 0, 1, 2, 3, 4, 5, 6, 7,
                                           8, 9, 10, 11, 12, 13, 14, 15);
}

static __device__ __forceinline__ v8f wmma_bf16(v16bf a, v16bf b, v8f c) {
    return __builtin_amdgcn_wmma_f32_16x16x32_bf16(
        /*neg_a=*/false, a, /*neg_b=*/false, b,
        /*c_mod=*/(short)0, c, /*reuse_a=*/false, /*reuse_b=*/false);
}

// ---------------------------------------------------------------------------
// CDNA5 async global->LDS DMA (ASYNCcnt-tracked), bypasses VGPRs entirely.
// IOFFSET is added to BOTH the LDS and the global address (ISA 08 §4.4).
// LDS address = low 32 bits of the generic pointer (aperture mapping).
// ---------------------------------------------------------------------------
static __device__ __forceinline__ unsigned lds_addr_of(const void* p) {
    return (unsigned)(size_t)p;
}

// copy 32 bytes (one thread) global -> LDS
static __device__ __forceinline__ void async_cp32(unsigned lds, const bf16_t* g) {
    asm volatile(
        "global_load_async_to_lds_b128 %0, %1, off\n\t"
        "global_load_async_to_lds_b128 %0, %1, off offset:16"
        :: "v"(lds), "v"(g) : "memory");
}

// copy 64 bytes (one thread) global -> LDS
static __device__ __forceinline__ void async_cp64(unsigned lds, const bf16_t* g) {
    asm volatile(
        "global_load_async_to_lds_b128 %0, %1, off\n\t"
        "global_load_async_to_lds_b128 %0, %1, off offset:16\n\t"
        "global_load_async_to_lds_b128 %0, %1, off offset:32\n\t"
        "global_load_async_to_lds_b128 %0, %1, off offset:48"
        :: "v"(lds), "v"(g) : "memory");
}

#define WAIT_ASYNC_4() asm volatile("s_wait_asynccnt 0x4" ::: "memory")
#define WAIT_ASYNC_0() asm volatile("s_wait_asynccnt 0x0" ::: "memory")

// ---------------------------------------------------------------------------
// Problem constants
// ---------------------------------------------------------------------------
#define BATCH     2
#define SEQ       2048
#define DMODEL    2048
#define NHEADS    16
#define NKV       4
#define HDIM      128
#define KVDIM     (NKV * HDIM)          // 512
#define MTOK      (BATCH * SEQ)         // 4096

// ---------------------------------------------------------------------------
// fp32 -> bf16 conversion
// ---------------------------------------------------------------------------
__global__ __launch_bounds__(256) void cvt_f32_bf16(const float* __restrict__ s,
                                                    bf16_t* __restrict__ d,
                                                    long n) {
    long idx = ((long)blockIdx.x * 256 + threadIdx.x) * 4;
    if (idx + 3 >= n) {
        for (long j = idx; j < n; ++j) d[j] = (bf16_t)s[j];
        return;
    }
    float4 v = *(const float4*)(s + idx);
    d[idx + 0] = (bf16_t)v.x;
    d[idx + 1] = (bf16_t)v.y;
    d[idx + 2] = (bf16_t)v.z;
    d[idx + 3] = (bf16_t)v.w;
}

// ---------------------------------------------------------------------------
// Tiled bf16 GEMM:  C[M,N] = A[M,K] * B[N,K]^T
// 128x128x32 tile, 256 threads (8 waves), each wave computes 32x64.
// Double-buffered LDS filled by GLOBAL_LOAD_ASYNC_TO_LDS_B128 (no VGPR
// staging, no spills). In-order ASYNCcnt => wait(4) retires exactly the
// older tile's 4 DMA ops while the next tile's 4 stay in flight.
// M,N multiples of 128; K multiple of 64.
// ---------------------------------------------------------------------------
#define GBM 128
#define GBN 128
#define GBK 32

static __device__ __forceinline__ void tile_compute(const bf16_t* As,
                                                    const bf16_t* Bs,
                                                    v8f acc[2][4], int wmi,
                                                    int wni, int fr, int khi) {
    v16bf af[2];
#pragma unroll
    for (int i = 0; i < 2; ++i) {
        const bf16_t* p = &As[(wmi * 32 + i * 16 + fr) * GBK + khi * 8];
        v8bf lo = *(const v8bf*)(p + khi * 8);
        v8bf hi = *(const v8bf*)(p + khi * 8 + 8);
        af[i] = combine8(lo, hi);
    }
    v16bf bfr[4];
#pragma unroll
    for (int j = 0; j < 4; ++j)
        bfr[j] = *(const v16bf*)(&Bs[(wni * 64 + j * 16 + fr) * GBK + khi * 16]);
#pragma unroll
    for (int i = 0; i < 2; ++i)
#pragma unroll
        for (int j = 0; j < 4; ++j)
            acc[i][j] = wmma_bf16(af[i], bfr[j], acc[i][j]);
}

__global__ __launch_bounds__(256) void gemm_bt_bf16(
    const bf16_t* __restrict__ A, const bf16_t* __restrict__ B,
    void* __restrict__ C, int M, int N, int K, int out_bf16) {
    __shared__ __align__(64) bf16_t As0[GBM * GBK];
    __shared__ __align__(64) bf16_t Bs0[GBN * GBK];
    __shared__ __align__(64) bf16_t As1[GBM * GBK];
    __shared__ __align__(64) bf16_t Bs1[GBN * GBK];

    const int tid  = threadIdx.x;
    const int lane = tid & 31;
    const int wave = tid >> 5;
    const int wmi  = wave & 3;   // 0..3 -> 32-row slab
    const int wni  = wave >> 2;  // 0..1 -> 64-col slab
    const int m0   = blockIdx.y * GBM;
    const int n0   = blockIdx.x * GBN;

    const int fr   = lane & 15;
    const int khi  = lane >> 4;

    const int lrow  = tid >> 1;        // 0..127  (LDS fill row)
    const int lhalf = (tid & 1) * 16;  // 0 / 16

    v8f acc[2][4];
#pragma unroll
    for (int i = 0; i < 2; ++i)
#pragma unroll
        for (int j = 0; j < 4; ++j) acc[i][j] = vzero8f();

    const bf16_t* ag = A + (size_t)(m0 + lrow) * K + lhalf;
    const bf16_t* bg = B + (size_t)(n0 + lrow) * K + lhalf;

    const unsigned la0 = lds_addr_of(&As0[lrow * GBK + lhalf]);
    const unsigned lb0 = lds_addr_of(&Bs0[lrow * GBK + lhalf]);
    const unsigned la1 = lds_addr_of(&As1[lrow * GBK + lhalf]);
    const unsigned lb1 = lds_addr_of(&Bs1[lrow * GBK + lhalf]);

    // prologue: DMA tile 0 into buffer 0 (4 async ops per thread)
    async_cp32(la0, ag);
    async_cp32(lb0, bg);

    for (int k0 = 0; k0 < K; k0 += 2 * GBK) {
        // DMA tile k0+GBK into buffer 1 (buf1 readers finished last iter)
        async_cp32(la1, ag + k0 + GBK);
        async_cp32(lb1, bg + k0 + GBK);
        WAIT_ASYNC_4();   // my 4 older ops (buf0 tile) have landed
        __syncthreads();  // everyone's buf0 portion landed
        tile_compute(As0, Bs0, acc, wmi, wni, fr, khi);
        __syncthreads();  // all waves done reading buf0 -> safe to overwrite

        if (k0 + 2 * GBK < K) {
            async_cp32(la0, ag + k0 + 2 * GBK);
            async_cp32(lb0, bg + k0 + 2 * GBK);
            WAIT_ASYNC_4();  // buf1 tile landed (older 4 of 8)
        } else {
            WAIT_ASYNC_0();  // drain: buf1 tile landed, nothing else in flight
        }
        __syncthreads();
        tile_compute(As1, Bs1, acc, wmi, wni, fr, khi);
        __syncthreads();  // buf1 readers done -> next iter may overwrite
    }
    WAIT_ASYNC_0();  // ensure no DMA outstanding at kernel end

    // store: C layout -> lane holds col fr, rows khi*8 + r
#pragma unroll
    for (int i = 0; i < 2; ++i) {
#pragma unroll
        for (int j = 0; j < 4; ++j) {
            const int row = m0 + wmi * 32 + i * 16 + khi * 8;
            const int col = n0 + wni * 64 + j * 16 + fr;
            if (out_bf16) {
                bf16_t* Cb = (bf16_t*)C;
#pragma unroll
                for (int r = 0; r < 8; ++r)
                    Cb[(size_t)(row + r) * N + col] = (bf16_t)acc[i][j][r];
            } else {
                float* Cf = (float*)C;
#pragma unroll
                for (int r = 0; r < 8; ++r)
                    Cf[(size_t)(row + r) * N + col] = acc[i][j][r];
            }
        }
    }
}

// ---------------------------------------------------------------------------
// In-place RMS-norm + RoPE (+ optional per-head gain) over bf16 [tok,heads,128]
// One wave per (token, head). 32 lanes x 4 elements = 128.
// ---------------------------------------------------------------------------
__global__ __launch_bounds__(256) void rms_rope_kernel(bf16_t* __restrict__ t,
                                                       const float* __restrict__ gain,
                                                       int nTokHead, int heads, int S) {
    const int wid = blockIdx.x * 8 + (threadIdx.x >> 5);
    if (wid >= nTokHead) return;
    const int lane  = threadIdx.x & 31;
    const int token = wid / heads;
    const int h     = wid - token * heads;
    const int s     = token & (S - 1);  // S power of two
    bf16_t* base = t + (size_t)wid * HDIM;

    const int i = lane * 2;  // pair index in [0,64)
    float a0 = (float)base[i],      a1 = (float)base[i + 1];
    float b0 = (float)base[i + 64], b1 = (float)base[i + 65];

    float ss = a0 * a0 + a1 * a1 + b0 * b0 + b1 * b1;
#pragma unroll
    for (int m = 1; m <= 16; m <<= 1) ss += __shfl_xor(ss, m, 32);
    const float inv = rsqrtf(ss * (1.0f / 128.0f) + 1.1920929e-7f);
    const float g = gain ? gain[h] : 1.0f;

#pragma unroll
    for (int k = 0; k < 2; ++k) {
        const int j = i + k;
        // inv_freq = 10000^(-j/64) = exp2(-j * log2(10000)/64)
        const float invf = exp2f(-(float)j * 0.2076205059304601f);
        const float ang  = (float)s * invf;
        float sn, cs;
        __sincosf(ang, &sn, &cs);
        const float x1 = (k ? a1 : a0) * inv;
        const float x2 = (k ? b1 : b0) * inv;
        base[j]      = (bf16_t)((x1 * cs + x2 * sn) * g);
        base[j + 64] = (bf16_t)((-x1 * sn + x2 * cs) * g);
    }
}

// ---------------------------------------------------------------------------
// Flash attention (causal, GQA r=4), bf16 WMMA + fp32 online softmax.
// 128 threads (4 waves); each wave owns a 16-row Q tile; key blocks of 32.
// K tile staged by async global->LDS DMA (overlaps the V transpose scatter);
// V goes through 16 registers because it must be transposed into LDS.
// ---------------------------------------------------------------------------
__global__ __launch_bounds__(128) void flash_attn_kernel(
    const bf16_t* __restrict__ Qb,   // [B,S,H,128]
    const bf16_t* __restrict__ Kb,   // [B,S,Hkv,128]
    const bf16_t* __restrict__ Vb,   // [B,S,Hkv,128]
    bf16_t* __restrict__ Yb,         // [B,S,H,128]
    int S) {
    __shared__ __align__(64) bf16_t lds_k[32 * HDIM];    // 8 KB
    __shared__ __align__(64) bf16_t lds_vt[HDIM * 32];   // 8 KB (transposed)
    __shared__ __align__(64) bf16_t p_lds[4 * 16 * 32];  // 4 KB

    const int b     = blockIdx.z;
    const int h     = blockIdx.y;
    const int wg_q0 = blockIdx.x * 64;
    const int tid   = threadIdx.x;
    const int lane  = tid & 31;
    const int wave  = tid >> 5;
    const int q0    = wg_q0 + wave * 16;
    const int hk    = h >> 2;  // GQA group of 4
    const int fr    = lane & 15;
    const int half  = lane >> 4;

    // --- Q fragments (16 rows x 128 hd = 4 K-chunks of 32) ---
    v16bf qf[4];
    {
        const bf16_t* qp = Qb + ((size_t)(b * S + q0 + fr) * NHEADS + h) * HDIM;
#pragma unroll
        for (int c = 0; c < 4; ++c) {
            v8bf lo = *(const v8bf*)(qp + c * 32 + half * 16);
            v8bf hi = *(const v8bf*)(qp + c * 32 + half * 16 + 8);
            qf[c] = combine8(lo, hi);
        }
    }

    v8f o[8];
#pragma unroll
    for (int c = 0; c < 8; ++c) o[c] = vzero8f();
    float Mr[8], Lr[8];
#pragma unroll
    for (int r = 0; r < 8; ++r) { Mr[r] = -3.0e38f; Lr[r] = 0.0f; }

    const float scale = 0.08838834764831845f;  // 1/sqrt(128)

    const int krow  = tid >> 2;        // 0..31 (kv row in tile)
    const int kpart = (tid & 3) * 32;  // hd quarter
    const unsigned lk = lds_addr_of(&lds_k[krow * HDIM + kpart]);

    for (int j0 = 0; j0 < wg_q0 + 64; j0 += 32) {
        // V into registers first (overlaps previous tile's compute)
        const size_t kvoff = ((size_t)(b * S + j0 + krow) * NKV + hk) * HDIM + kpart;
        const uint4* vs = (const uint4*)(Vb + kvoff);
        uint4 vr0 = vs[0], vr1 = vs[1], vr2 = vs[2], vr3 = vs[3];

        __syncthreads();  // previous tile's LDS readers are done

        // K tile: async DMA straight into LDS (64B per thread)
        async_cp64(lk, Kb + kvoff);

        // V tile: transpose into LDS from registers
        {
            bf16_t vtmp[32];
            *(uint4*)&vtmp[0]  = vr0;
            *(uint4*)&vtmp[8]  = vr1;
            *(uint4*)&vtmp[16] = vr2;
            *(uint4*)&vtmp[24] = vr3;
#pragma unroll
            for (int i = 0; i < 32; ++i)
                lds_vt[(kpart + i) * 32 + krow] = vtmp[i];
        }
        WAIT_ASYNC_0();   // my K DMA landed
        __syncthreads();  // everyone's K + V staged

        if (j0 <= q0 + 15) {
            // scores: two 16x16 tiles (keys j0..+15, j0+16..+31)
            v8f sc[2];
#pragma unroll
            for (int t = 0; t < 2; ++t) {
                sc[t] = vzero8f();
#pragma unroll
                for (int c = 0; c < 4; ++c) {
                    v16bf kfrag = *(const v16bf*)(
                        &lds_k[(t * 16 + fr) * HDIM + c * 32 + half * 16]);
                    sc[t] = wmma_bf16(qf[c], kfrag, sc[t]);
                }
            }
            // masked scores + online softmax (fp32)
            float s0[8], s1[8], tmax[8];
#pragma unroll
            for (int r = 0; r < 8; ++r) {
                const int mrow = half * 8 + r;
                float a = sc[0][r] * scale;
                float c2 = sc[1][r] * scale;
                if (j0 + fr > q0 + mrow) a = -1.0e30f;
                if (j0 + 16 + fr > q0 + mrow) c2 = -1.0e30f;
                s0[r] = a; s1[r] = c2;
                tmax[r] = fmaxf(a, c2);
            }
#pragma unroll
            for (int m = 1; m <= 8; m <<= 1)
#pragma unroll
                for (int r = 0; r < 8; ++r)
                    tmax[r] = fmaxf(tmax[r], __shfl_xor(tmax[r], m, 32));

            float p0[8], p1[8], rs[8], sclr[8];
#pragma unroll
            for (int r = 0; r < 8; ++r) {
                const float nm = fmaxf(Mr[r], tmax[r]);
                sclr[r] = __expf(Mr[r] - nm);
                Mr[r] = nm;
                p0[r] = __expf(s0[r] - nm);
                p1[r] = __expf(s1[r] - nm);
                rs[r] = p0[r] + p1[r];
            }
#pragma unroll
            for (int m = 1; m <= 8; m <<= 1)
#pragma unroll
                for (int r = 0; r < 8; ++r)
                    rs[r] += __shfl_xor(rs[r], m, 32);
#pragma unroll
            for (int r = 0; r < 8; ++r) Lr[r] = Lr[r] * sclr[r] + rs[r];
#pragma unroll
            for (int c = 0; c < 8; ++c)
#pragma unroll
                for (int r = 0; r < 8; ++r) o[c][r] *= sclr[r];

            // relayout P (C-layout) -> A-layout via per-wave LDS scratch
            bf16_t* pw = &p_lds[wave * 512];
#pragma unroll
            for (int r = 0; r < 8; ++r) {
                const int mrow = half * 8 + r;
                pw[mrow * 32 + fr]      = (bf16_t)p0[r];
                pw[mrow * 32 + 16 + fr] = (bf16_t)p1[r];
            }
            asm volatile("s_wait_dscnt 0x0" ::: "memory");  // per-wave LDS RAW
            v16bf pf;
            {
                const bf16_t* pp = &p_lds[wave * 512 + fr * 32];
                v8bf lo = *(const v8bf*)(pp + half * 16);
                v8bf hi = *(const v8bf*)(pp + half * 16 + 8);
                pf = combine8(lo, hi);
            }
            // O += P(16x32) * V(32x128)
#pragma unroll
            for (int c = 0; c < 8; ++c) {
                v16bf vfrag = *(const v16bf*)(
                    &lds_vt[(c * 16 + fr) * 32 + half * 16]);
                o[c] = wmma_bf16(pf, vfrag, o[c]);
            }
        }
    }

    // epilogue: divide by row sums, store bf16
    float rinv[8];
#pragma unroll
    for (int r = 0; r < 8; ++r) rinv[r] = 1.0f / Lr[r];
#pragma unroll
    for (int c = 0; c < 8; ++c) {
#pragma unroll
        for (int r = 0; r < 8; ++r) {
            const int mrow = half * 8 + r;
            Yb[((size_t)(b * S + q0 + mrow) * NHEADS + h) * HDIM + c * 16 + fr] =
                (bf16_t)(o[c][r] * rinv[r]);
        }
    }
}

// ---------------------------------------------------------------------------
// Host orchestration
// ---------------------------------------------------------------------------
extern "C" void kernel_launch(void* const* d_in, const int* in_sizes, int n_in,
                              void* d_out, int out_size, void* d_ws, size_t ws_size,
                              hipStream_t stream) {
    const float* x     = (const float*)d_in[0];  // [B,S,D]
    const float* Wq    = (const float*)d_in[1];  // [D,D]
    const float* Wk    = (const float*)d_in[2];  // [KVDIM,D]
    const float* Wv    = (const float*)d_in[3];  // [KVDIM,D]
    const float* Wproj = (const float*)d_in[4];  // [D,D]
    const float* qgain = (const float*)d_in[5];  // [H]
    float* out = (float*)d_out;

    char* w = (char*)d_ws;
    bf16_t* xb  = (bf16_t*)w;                     w += (size_t)MTOK * DMODEL * 2;
    bf16_t* wqb = (bf16_t*)w;                     w += (size_t)DMODEL * DMODEL * 2;
    bf16_t* wkb = (bf16_t*)w;                     w += (size_t)KVDIM * DMODEL * 2;
    bf16_t* wvb = (bf16_t*)w;                     w += (size_t)KVDIM * DMODEL * 2;
    bf16_t* wpb = (bf16_t*)w;                     w += (size_t)DMODEL * DMODEL * 2;
    bf16_t* qb  = (bf16_t*)w;                     w += (size_t)MTOK * DMODEL * 2;
    bf16_t* kb  = (bf16_t*)w;                     w += (size_t)MTOK * KVDIM * 2;
    bf16_t* vb  = (bf16_t*)w;                     w += (size_t)MTOK * KVDIM * 2;
    bf16_t* yb  = (bf16_t*)w;                     w += (size_t)MTOK * DMODEL * 2;

    // 1) fp32 -> bf16
    const long nx = (long)MTOK * DMODEL;
    const long nw = (long)DMODEL * DMODEL;
    const long nk = (long)KVDIM * DMODEL;
    cvt_f32_bf16<<<dim3(nx / 1024), dim3(256), 0, stream>>>(x, xb, nx);
    cvt_f32_bf16<<<dim3(nw / 1024), dim3(256), 0, stream>>>(Wq, wqb, nw);
    cvt_f32_bf16<<<dim3(nk / 1024), dim3(256), 0, stream>>>(Wk, wkb, nk);
    cvt_f32_bf16<<<dim3(nk / 1024), dim3(256), 0, stream>>>(Wv, wvb, nk);
    cvt_f32_bf16<<<dim3(nw / 1024), dim3(256), 0, stream>>>(Wproj, wpb, nw);

    // 2) QKV projections (bf16 out)
    gemm_bt_bf16<<<dim3(DMODEL / GBN, MTOK / GBM), dim3(256), 0, stream>>>(
        xb, wqb, qb, MTOK, DMODEL, DMODEL, 1);
    gemm_bt_bf16<<<dim3(KVDIM / GBN, MTOK / GBM), dim3(256), 0, stream>>>(
        xb, wkb, kb, MTOK, KVDIM, DMODEL, 1);
    gemm_bt_bf16<<<dim3(KVDIM / GBN, MTOK / GBM), dim3(256), 0, stream>>>(
        xb, wvb, vb, MTOK, KVDIM, DMODEL, 1);

    // 3) RMS-norm + RoPE (+gain for Q), in place
    rms_rope_kernel<<<dim3((MTOK * NHEADS) / 8), dim3(256), 0, stream>>>(
        qb, qgain, MTOK * NHEADS, NHEADS, SEQ);
    rms_rope_kernel<<<dim3((MTOK * NKV) / 8), dim3(256), 0, stream>>>(
        kb, nullptr, MTOK * NKV, NKV, SEQ);

    // 4) causal flash attention
    flash_attn_kernel<<<dim3(SEQ / 64, NHEADS, BATCH), dim3(128), 0, stream>>>(
        qb, kb, vb, yb, SEQ);

    // 5) output projection (fp32 out)
    gemm_bt_bf16<<<dim3(DMODEL / GBN, MTOK / GBM), dim3(256), 0, stream>>>(
        yb, wpb, out, MTOK, DMODEL, DMODEL, 0);
}
